// ClassConfusionLoss_54417235640824
// MI455X (gfx1250) — compile-verified, matrix operand order checked by reading, not run on hardware
//
#include <hip/hip_runtime.h>
#include <hip/hip_bf16.h>

typedef __attribute__((ext_vector_type(16))) _Float16 v16h;
typedef __attribute__((ext_vector_type(4)))  _Float16 v4h;
typedef __attribute__((ext_vector_type(8)))  float    v8f;

#define EPSF 1e-12f
#define CCH   64          // channels (== batch)
#define WH    16384       // 128*128 spatial
#define NTILE 4096        // 64 batches * 64 s-blocks of 256 pixels
#define NWG   512         // workgroups for main pass
#define KPITCH 36         // padded K pitch (halves) for conflict-free LDS reads
#define GSTRIDE (64*KPITCH) // 2304 halves per 32-pixel group
#define YBYTES (8*GSTRIDE*2) // 36864 B for the f16 x-tile

// ---- Kernel 1: rden[b][s] = 1 / (sum_c pred[b,c,s] + eps), float4-wide ----
__global__ void k_denom(const float* __restrict__ pred, float* __restrict__ rden) {
    int idx4 = blockIdx.x * 256 + threadIdx.x;         // 0 .. 262143 (quads)
    int b  = idx4 >> 12;                               // 4096 quads per batch
    int sq = idx4 & 4095;
    const float4* p = (const float4*)(pred + ((size_t)b << 20)) + sq;
    float4 a = make_float4(0.f, 0.f, 0.f, 0.f);
#pragma unroll 8
    for (int c = 0; c < CCH; ++c) {
        float4 v = p[(size_t)c << 12];                 // row stride 4096 quads
        a.x += v.x; a.y += v.y; a.z += v.z; a.w += v.w;
    }
    float4 r = make_float4(1.f / (a.x + EPSF), 1.f / (a.y + EPSF),
                           1.f / (a.z + EPSF), 1.f / (a.w + EPSF));
    ((float4*)rden)[idx4] = r;
}

// ------ Kernel 2: fused normalize/entropy/weight + f16 WMMA Gram -----------
__global__ void __launch_bounds__(256) k_main(const float* __restrict__ pred,
                                              const int*   __restrict__ gt,
                                              const float* __restrict__ rden,
                                              float* __restrict__ partials) {
    __shared__ __align__(16) unsigned char smemraw[YBYTES + 512];
    _Float16* yls = (_Float16*)smemraw;                // unscaled x, f16
    _Float16* scl = (_Float16*)(smemraw + YBYTES);     // per-pixel sqrt(n*wraw)

    const int tid = threadIdx.x;
    const int l   = tid & 31;          // lane
    const int wid = tid >> 5;          // wave id 0..7
    const int m   = l & 15;            // N / M position within 16
    const int hi  = l >> 4;            // half-wave select
    const int koff = hi << 3;          // K offset: 0 or 8

    v8f acc[4][4];
#pragma unroll
    for (int ci = 0; ci < 4; ++ci)
#pragma unroll
        for (int kj = 0; kj < 4; ++kj) acc[ci][kj] = (v8f)0.f;

    for (int tile = blockIdx.x; tile < NTILE; tile += NWG) {
        // ---- prefetch next tile's streams into L2 while we compute ----
        if (tile + NWG < NTILE) {
            const int nt = tile + NWG;
            const int nb = nt >> 6, ns0 = (nt & 63) << 8;
            const float* np = pred + ((size_t)nb << 20) + ns0 + tid;
            const int*   ng = gt   + ((size_t)nb << 20) + ns0 + tid;
#pragma unroll
            for (int c = 0; c < CCH; c += 16) {
                __builtin_prefetch(np + ((size_t)c << 14), 0, 3);
                __builtin_prefetch(ng + ((size_t)c << 14), 0, 3);
            }
        }

        const int b  = tile >> 6;
        const int s0 = (tile & 63) << 8;                 // 256-pixel s-block
        // ---- Phase A: each thread owns one pixel of the tile ----
        const size_t gbase = ((size_t)b << 20) + s0 + tid;   // c=0 element
        const int dbase = s0 + tid;                          // rden index, c=0
        _Float16* col = yls + (tid >> 5) * GSTRIDE + (tid & 31); // + c*KPITCH
        float entacc = 0.f;
        int   n = 0;
#pragma unroll 4
        for (int c = 0; c < CCH; ++c) {
            float pv = pred[gbase + ((size_t)c << 14)];
            float rv = rden[dbase + (c << 14)];
            float x  = pv * rv;                             // pred/(sum+eps)
            entacc   = fmaf(x, __logf(x + EPSF), entacc);   // = -ent
            n       += (gt[gbase + ((size_t)c << 14)] == 1);
            col[c * KPITCH] = (_Float16)x;                  // unscaled
        }
        // w_raw = 1 + exp(-ent) = 1 + exp(entacc);  weight s = n*w_raw >= 0
        scl[(tid >> 5) * 32 + (tid & 31)] =
            (_Float16)sqrtf((float)n * (1.0f + __expf(entacc)));
        __syncthreads();

        // ---- Phase B: wave `wid` consumes pixel-group `wid` (K = 32) ----
        // scale fragment: same per-lane K mapping as data fragments
        union { v4h q4[4]; v16h v; } us;
        const _Float16* sp = scl + wid * 32 + koff;
        us.q4[0] = *(const v4h*)(sp + 0);
        us.q4[1] = *(const v4h*)(sp + 4);
        us.q4[2] = *(const v4h*)(sp + 16);
        us.q4[3] = *(const v4h*)(sp + 20);
        const v16h sv = us.v;

        const _Float16* gb = yls + wid * GSTRIDE + koff;
        v16h fr[4];
#pragma unroll
        for (int q = 0; q < 4; ++q) {
            const _Float16* p = gb + (q * 16 + m) * KPITCH;
            union { v4h q4[4]; v16h v; } u;
            u.q4[0] = *(const v4h*)(p + 0);   // K koff+0..3
            u.q4[1] = *(const v4h*)(p + 4);   // K koff+4..7
            u.q4[2] = *(const v4h*)(p + 16);  // K koff+16..19
            u.q4[3] = *(const v4h*)(p + 20);  // K koff+20..23
            fr[q] = u.v * sv;                 // v_pk_mul_f16: y = sqrt(s)*x
        }
#pragma unroll
        for (int ci = 0; ci < 4; ++ci)
#pragma unroll
            for (int kj = 0; kj < 4; ++kj)
                acc[ci][kj] = __builtin_amdgcn_wmma_f32_16x16x32_f16(
                    false, fr[ci], false, fr[kj],
                    (short)0, acc[ci][kj], false, false);
        __syncthreads();
    }

    // ---- deterministic in-WG reduction: waves add in fixed order ----
    float* covls = (float*)smemraw;                      // reuse 16 KB of LDS
    for (int e = tid; e < 4096; e += 256) covls[e] = 0.f;
    __syncthreads();
    for (int w = 0; w < 8; ++w) {
        if (wid == w) {
#pragma unroll
            for (int ci = 0; ci < 4; ++ci)
#pragma unroll
                for (int kj = 0; kj < 4; ++kj)
#pragma unroll
                    for (int r = 0; r < 8; ++r) {
                        int row = ci * 16 + r + (hi << 3);
                        int colc = kj * 16 + m;
                        covls[row * 64 + colc] += acc[ci][kj][r];
                    }
        }
        __syncthreads();
    }
    for (int e = tid; e < 4096; e += 256)
        partials[(size_t)blockIdx.x * 4096 + e] = covls[e];
}

// ---------------- Kernel 3: reduce partials over workgroups ----------------
__global__ void k_reduce(const float* __restrict__ partials, float* __restrict__ cov) {
    int e = blockIdx.x * 256 + threadIdx.x;              // 16 blocks -> 4096
    float s = 0.f;
    for (int w = 0; w < NWG; ++w) s += partials[(size_t)w * 4096 + e];
    cov[e] = s;
}

// -------- Kernel 4: row-normalize, (sum - trace)/C, single scalar ----------
__global__ void k_final(const float* __restrict__ cov, float* __restrict__ out) {
    __shared__ float covls[4096];
    __shared__ float rows[64];
    __shared__ float red[256];
    const int tid = threadIdx.x;
    for (int e = tid; e < 4096; e += 256) covls[e] = cov[e];
    __syncthreads();
    if (tid < 64) {
        float s = 0.f;
        for (int j = 0; j < 64; ++j) s += covls[tid * 64 + j];
        rows[tid] = s;                                   // rs[k] = sum_j cov[k,j]
    }
    __syncthreads();
    float part = 0.f;
    for (int e = tid; e < 4096; e += 256) {
        int c = e >> 6, k = e & 63;
        float v = covls[e] / rows[k];                    // cov[c,k]/rs[k]
        if (c != k) part += v;                           // sum - trace
    }
    red[tid] = part;
    __syncthreads();
    for (int off = 128; off > 0; off >>= 1) {
        if (tid < off) red[tid] += red[tid + off];
        __syncthreads();
    }
    if (tid == 0) out[0] = red[0] / 64.0f;
}

extern "C" void kernel_launch(void* const* d_in, const int* in_sizes, int n_in,
                              void* d_out, int out_size, void* d_ws, size_t ws_size,
                              hipStream_t stream) {
    (void)in_sizes; (void)n_in; (void)out_size; (void)ws_size;
    const float* pred = (const float*)d_in[0];
    const int*   gt   = (const int*)d_in[1];
    float* ws       = (float*)d_ws;
    float* rden     = ws;                                  // 1M floats (4 MB)
    float* partials = ws + (1 << 20);                      // 512*4096 (8 MB)
    float* cov      = ws + (1 << 20) + (NWG * 4096);       // 4096 floats

    k_denom <<<1024, 256, 0, stream>>>(pred, rden);
    k_main  <<<NWG,  256, 0, stream>>>(pred, gt, rden, partials);
    k_reduce<<<16,   256, 0, stream>>>(partials, cov);
    k_final <<<1,    256, 0, stream>>>(cov, (float*)d_out);
}
